// AdvancedWaveletDecomp_80985903333578
// MI455X (gfx1250) — matrix operand order repaired and checked by reading, not compiled, not run on hardware
//
#include <hip/hip_runtime.h>
#include <hip/hip_bf16.h>
#include <math.h>

// ---------------------------------------------------------------------------
// AdvancedWaveletDecomp for MI455X (gfx1250, wave32, WMMA).
// Heavy path: fused neural-ODE conv pairs as bf16 WMMA GEMMs (f32 accum),
// RK4 AXPY updates fused into the conv2 epilogue (k never materialized).
// ---------------------------------------------------------------------------

typedef __attribute__((ext_vector_type(16))) __bf16 v16bf;
typedef __attribute__((ext_vector_type(8)))  float  v8f;

#define B_    32
#define C_    64
#define L_    4096
#define NTOT  (B_*C_*L_)          // 8,388,608
#define TILE  256                 // columns of L per block
#define FL_   8
#define MID_  16
#define REG_  0.01f
#define ODE_H 0.1f

// ---------------- device helpers ----------------

__device__ __forceinline__ float gelu_f(float x){
  return 0.5f * x * (1.0f + erff(x * 0.70710678118654752440f));
}
__device__ __forceinline__ float sigm_f(float x){
  return 1.0f / (1.0f + __expf(-x));
}

// A fragment (16x32 bf16): W is 64x64 row-major [out_ch][in_ch] in LDS.
// lanes 0-15: row=lane,    K = ks*32 + {0..7, 16..23}
// lanes16-31: row=lane-16, K = ks*32 + {8..15,24..31}
__device__ __forceinline__ v16bf frag_a(const __bf16* W, int mtile, int ks, int lane){
  const int row = mtile*16 + (lane & 15);
  const int kb  = ks*32 + ((lane >> 4) << 3);
  const __bf16* p = W + row*64 + kb;
  union { v16bf v; uint4 u[2]; } r;
  r.u[0] = *reinterpret_cast<const uint4*>(p);        // K = kb .. kb+7
  r.u[1] = *reinterpret_cast<const uint4*>(p + 16);   // K = kb+16 .. kb+23
  return r.v;
}

// B fragment (32x16 bf16): ACT is [col][64] bf16 in LDS (channel-contiguous).
// lanes 0-15: col=base+lane,    K = ks*32 + {0..7, 16..23}
// lanes16-31: col=base+lane-16, K = ks*32 + {8..15,24..31}
__device__ __forceinline__ v16bf frag_b(const __bf16* ACT, int colbase, int ks, int lane){
  const int col = colbase + (lane & 15);
  const int cb  = ks*32 + ((lane >> 4) << 3);
  const __bf16* p = ACT + col*64 + cb;
  union { v16bf v; uint4 u[2]; } r;
  r.u[0] = *reinterpret_cast<const uint4*>(p);        // K = cb .. cb+7
  r.u[1] = *reinterpret_cast<const uint4*>(p + 16);   // K = cb+16 .. cb+23
  return r.v;
}

__device__ __forceinline__ v8f wmma_bf16(v16bf a, v16bf b, v8f c){
  return __builtin_amdgcn_wmma_f32_16x16x32_bf16(false, a, false, b, (short)0, c, false, false);
}

// ---------------- fused ODE stage ------------------------------------------
// k = f(zin) = tanh(conv2(gelu(conv1(zin)))) - damp*zin, then RK4 update:
//   mode 0: acc = zst + wa*k ; zo = zst + wt*k        (first stage)
//   mode 1: acc += wa*k      ; zo = zst + wt*k        (middle stages)
//   mode 2: zo = acc + wa*k                           (last stage -> new z)
// grid = B * (L/TILE) = 512 blocks, 256 threads (8 waves).
// dynamic LDS: sW1[3*64*64] | sW2[3*64*64] | sZ[288*64] | sH[272*64]  (bf16)

#define FE_W1  0
#define FE_W2  (3*64*64)
#define FE_Z   (2*3*64*64)
#define FE_H   (2*3*64*64 + 288*64)
#define FE_LDS ((2*3*64*64 + 288*64 + 272*64) * 2)   // 120,832 bytes

__global__ __launch_bounds__(256)
void k_f_eval(const float* __restrict__ zin, const float* __restrict__ zst,
              float* __restrict__ accb, float* __restrict__ zo,
              const float* __restrict__ w1, const float* __restrict__ b1,
              const float* __restrict__ w2, const float* __restrict__ b2,
              const float* __restrict__ damp_p,
              float wa, float wt, int mode)
{
  extern __shared__ __bf16 smem[];
  __bf16* sW1 = smem + FE_W1;
  __bf16* sW2 = smem + FE_W2;
  __bf16* sZ  = smem + FE_Z;    // [288][64], col 0 <-> l0-2, edge-clamped
  __bf16* sH  = smem + FE_H;    // [272][64], col 0 <-> l0-1

  const int tid  = threadIdx.x, lane = tid & 31, wave = tid >> 5;
  const int b    = blockIdx.x >> 4;
  const int l0   = (blockIdx.x & 15) * TILE;
  const float damp = damp_p[0];

  // early prefetch of this block's z tile (global_prefetch_b8 path)
  {
    const int r = tid & 63, seg = tid >> 6;     // 4 x 64-col segments
    __builtin_prefetch(&zin[(b*64 + r)*L_ + l0 + seg*64], 0, 0);
  }

  // stage weights (HBM layout: [o][c][t] f32) -> LDS [t][o*64+c] bf16
  for (int i = tid; i < 3*64*64; i += 256){
    const int t = i % 3, oc = i / 3;
    sW1[t*4096 + oc] = (__bf16)w1[i];
    sW2[t*4096 + oc] = (__bf16)w2[i];
  }
  // stage z tile (+halo 2, edge clamp), channel-contiguous bf16
  for (int i = tid; i < 288*64; i += 256){
    const int c = i / 288, cc = i % 288;
    int gl = l0 + cc - 2;
    gl = gl < 0 ? 0 : (gl > L_-1 ? L_-1 : gl);
    sZ[cc*64 + c] = (__bf16)zin[(b*64 + c)*L_ + gl];
  }
  __syncthreads();

  // ---- conv1 (64->64, k=3) + GELU over 272 cols (258 used) ----
  {
    const int mtile = wave & 3;
    v16bf afr[6];
    #pragma unroll
    for (int u = 0; u < 6; ++u) afr[u] = frag_a(sW1 + (u >> 1)*4096, mtile, u & 1, lane);
    for (int nt = (wave >> 2); nt < 17; nt += 2){
      v16bf bfr[6];
      #pragma unroll
      for (int u = 0; u < 6; ++u) bfr[u] = frag_b(sZ, nt*16 + (u >> 1), u & 1, lane);
      v8f acc = {0.f,0.f,0.f,0.f,0.f,0.f,0.f,0.f};
      #pragma unroll
      for (int u = 0; u < 6; ++u) acc = wmma_bf16(afr[u], bfr[u], acc);
      const int j  = nt*16 + (lane & 15);
      const int mb = mtile*16 + ((lane >> 4) << 3);
      #pragma unroll
      for (int e = 0; e < 8; ++e){
        const int m = mb + e;
        sH[j*64 + m] = (__bf16)gelu_f(acc[e] + b1[m]);
      }
    }
  }
  __syncthreads();

  // ---- conv2 (64->64, k=3) + tanh - damp*z + fused RK4 update ----
  {
    const int mtile = wave & 3;
    v16bf afr[6];
    #pragma unroll
    for (int u = 0; u < 6; ++u) afr[u] = frag_a(sW2 + (u >> 1)*4096, mtile, u & 1, lane);
    for (int nt = (wave >> 2); nt < 16; nt += 2){
      v16bf bfr[6];
      #pragma unroll
      for (int u = 0; u < 6; ++u) bfr[u] = frag_b(sH, nt*16 + (u >> 1), u & 1, lane);
      v8f acc = {0.f,0.f,0.f,0.f,0.f,0.f,0.f,0.f};
      #pragma unroll
      for (int u = 0; u < 6; ++u) acc = wmma_bf16(afr[u], bfr[u], acc);
      const int j  = nt*16 + (lane & 15);
      const int mb = mtile*16 + ((lane >> 4) << 3);
      #pragma unroll
      for (int e = 0; e < 8; ++e){
        const int m   = mb + e;
        const int idx = (b*64 + m)*L_ + l0 + j;
        const float kval = tanhf(acc[e] + b2[m]) - damp * zin[idx];
        if (mode == 0){
          const float zv = zst[idx];
          accb[idx] = zv + wa*kval;
          zo[idx]   = zv + wt*kval;
        } else if (mode == 1){
          const float zv = zst[idx];
          accb[idx] += wa*kval;
          zo[idx]   = zv + wt*kval;
        } else {
          zo[idx] = accb[idx] + wa*kval;
        }
      }
    }
  }
}

// ---------------- per-(b,c) mean over L ----------------

__global__ __launch_bounds__(256)
void k_mean(const float* __restrict__ ap, float* __restrict__ means)
{
  __shared__ float red[256];
  const int row = blockIdx.x;            // b*64 + c
  float s = 0.f;
  for (int l = threadIdx.x; l < L_; l += 256) s += ap[row*L_ + l];
  red[threadIdx.x] = s; __syncthreads();
  for (int off = 128; off; off >>= 1){
    if (threadIdx.x < off) red[threadIdx.x] += red[threadIdx.x + off];
    __syncthreads();
  }
  if (threadIdx.x == 0) means[row] = red[0] * (1.0f / L_);
}

// ---------------- dywan: per-batch MLP + ortho regularizer ----------------
// shift simplifies: roll is a permutation, so
//   sum_s sum_{i,j} |lo_n_i * lo_n_{(j-s)%8}| = 3 * (sum|lo_n|)^2

__global__ __launch_bounds__(256)
void k_dywan(const float* __restrict__ means,
             const float* __restrict__ sw,  const float* __restrict__ sb,
             const float* __restrict__ g1w, const float* __restrict__ g1b,
             const float* __restrict__ g2w, const float* __restrict__ g2b,
             float* __restrict__ lo_out, float* __restrict__ hi_out,
             float* __restrict__ ortho, int lev)
{
  __shared__ float red[B_];
  const int tid = threadIdx.x;
  if (tid < B_){
    const float* mb = means + tid*64;
    float stat[64];
    for (int i = 0; i < 64; ++i){
      float s = sb[i];
      for (int c = 0; c < 64; ++c) s += mb[c] * sw[i*64 + c];
      stat[i] = gelu_f(s);
    }
    float g1[128];
    for (int j = 0; j < 128; ++j){
      float s = g1b[j];
      for (int i = 0; i < 64; ++i) s += stat[i] * g1w[j*64 + i];
      g1[j] = gelu_f(s);
    }
    float g[16];
    for (int j = 0; j < 16; ++j){
      float s = g2b[j];
      for (int i = 0; i < 128; ++i) s += g1[i] * g2w[j*128 + i];
      g[j] = s;
    }
    for (int t = 0; t < FL_; ++t){ lo_out[tid*FL_+t] = g[t]; hi_out[tid*FL_+t] = g[FL_+t]; }

    float lp[10]; lp[0] = 0.f; lp[9] = 0.f;
    for (int t = 0; t < 8; ++t) lp[t+1] = g[t];
    float smooth = 0.f;
    for (int d = 0; d < 9; ++d) smooth += fabsf(lp[d+1] - lp[d]);
    float nrm = 0.f;
    for (int t = 0; t < 8; ++t) nrm += g[t]*g[t];
    const float inv = 1.0f / (sqrtf(nrm) + 1e-8f);
    float s1 = 0.f, s2 = 0.f;
    for (int t = 0; t < 8; ++t){ const float v = g[t]*inv; s1 += fabsf(v); s2 += v*v; }
    const float shift_b = 3.0f * s1 * s1;
    const float amp_b   = fabsf(s2 - 1.0f);
    red[tid] = REG_*(shift_b/(B_*64.0f) + amp_b/(float)B_) + 0.1f*(smooth/(B_*9.0f));
  }
  __syncthreads();
  if (tid == 0){
    float s = 0.f;
    for (int i = 0; i < B_; ++i) s += red[i];
    if (lev == 0) ortho[0] = s; else ortho[0] += s;
  }
}

// ---------------- 8-tap FIR (edge pad): new_approx & detail ----------------
// LDS-tiled to kill the 8x tap read amplification.

__global__ __launch_bounds__(256)
void k_fir(const float* __restrict__ ap, float* __restrict__ nap,
           float* __restrict__ det,
           const float* __restrict__ lo, const float* __restrict__ hi)
{
  __shared__ float s[TILE + 8];          // col 0 <-> l0-4
  const int row = blockIdx.x >> 4;       // b*64 + c
  const int l0  = (blockIdx.x & 15) * TILE;
  const int b   = row >> 6;
  const int tid = threadIdx.x;

  for (int i = tid; i < TILE + 8; i += 256){
    int gl = l0 + i - 4;                 // pad_l = FL/2 = 4 (edge clamp)
    gl = gl < 0 ? 0 : (gl > L_-1 ? L_-1 : gl);
    s[i] = ap[row*L_ + gl];
  }
  __syncthreads();

  float slo = 0.f, shi = 0.f;
  #pragma unroll
  for (int t = 0; t < 8; ++t){
    const float v = s[tid + t];
    slo += v * lo[b*8 + t];
    shi += v * hi[b*8 + t];
  }
  const int idx = row*L_ + l0 + tid;
  nap[idx] = slo;
  det[idx] = shi;
}

// ---------------- reconstruction level (attention + WMMA gate conv) -------
// dynamic LDS: sGW[3*64*64] | sC[264*64] | sMid[256*16]  (bf16)

#define RC_GW  0
#define RC_C   (3*64*64)
#define RC_MID (3*64*64 + 264*64)
#define RC_LDS ((3*64*64 + 264*64 + 256*16) * 2)     // 66,560 bytes

__global__ __launch_bounds__(256)
void k_recon(const float* __restrict__ cur_in, float* __restrict__ cur_out,
             float* __restrict__ det,
             const float* __restrict__ gw,  const float* __restrict__ gb,
             const float* __restrict__ aw1, const float* __restrict__ ab1,
             const float* __restrict__ aw2, const float* __restrict__ ab2,
             int has_att)
{
  extern __shared__ __bf16 smem[];
  __bf16* sGW  = smem + RC_GW;
  __bf16* sC   = smem + RC_C;     // [264][64], col 0 <-> l0-1, zero pad
  __bf16* sMid = smem + RC_MID;   // [256][16]

  const int tid  = threadIdx.x, lane = tid & 31, wave = tid >> 5;
  const int b    = blockIdx.x >> 4;
  const int l0   = (blockIdx.x & 15) * TILE;

  {
    const int r = tid & 63, seg = tid >> 6;
    __builtin_prefetch(&cur_in[(b*64 + r)*L_ + l0 + seg*64], 0, 0);
  }

  for (int i = tid; i < 3*64*64; i += 256){
    const int t = i % 3, oc = i / 3;
    sGW[t*4096 + oc] = (__bf16)gw[i];
  }
  for (int i = tid; i < 264*64; i += 256){
    const int c = i / 264, cc = i % 264;
    const int gl = l0 + cc - 1;
    sC[cc*64 + c] = (__bf16)((gl >= 0 && gl < L_) ? cur_in[(b*64 + c)*L_ + gl] : 0.0f);
  }
  __syncthreads();

  if (has_att){
    for (int i = tid; i < 256*16; i += 256){
      const int j = i >> 4, m = i & 15;
      float s = ab1[m];
      for (int c = 0; c < 64; ++c) s += (float)sC[(j+1)*64 + c] * aw1[m*64 + c];
      sMid[j*16 + m] = (__bf16)gelu_f(s);
    }
  }
  __syncthreads();

  const int mtile = wave & 3;
  v16bf afr[6];
  #pragma unroll
  for (int u = 0; u < 6; ++u) afr[u] = frag_a(sGW + (u >> 1)*4096, mtile, u & 1, lane);
  for (int nt = (wave >> 2); nt < 16; nt += 2){
    v16bf bfr[6];
    #pragma unroll
    for (int u = 0; u < 6; ++u) bfr[u] = frag_b(sC, nt*16 + (u >> 1), u & 1, lane);
    v8f acc = {0.f,0.f,0.f,0.f,0.f,0.f,0.f,0.f};
    #pragma unroll
    for (int u = 0; u < 6; ++u) acc = wmma_bf16(afr[u], bfr[u], acc);

    const int j  = nt*16 + (lane & 15);
    const int mb = mtile*16 + ((lane >> 4) << 3);
    #pragma unroll
    for (int e = 0; e < 8; ++e){
      const int m   = mb + e;
      const int idx = (b*64 + m)*L_ + l0 + j;
      const float gate = sigm_f(acc[e] + gb[m]);
      float d = det[idx];
      if (has_att){
        float s = ab2[m];
        #pragma unroll
        for (int q = 0; q < 16; ++q) s += (float)sMid[j*16 + q] * aw2[m*16 + q];
        d += d * sigm_f(s);
      }
      cur_out[idx] = cur_in[idx] + gate * d;
      det[idx] = d;
    }
  }
}

// ---------------------------------------------------------------------------
// host side
// ---------------------------------------------------------------------------

extern "C" void kernel_launch(void* const* d_in, const int* in_sizes, int n_in,
                              void* d_out, int out_size, void* d_ws, size_t ws_size,
                              hipStream_t stream)
{
  (void)in_sizes; (void)n_in; (void)out_size; (void)ws_size;

  const float* x    = (const float*)d_in[0];
  const float* c1w  = (const float*)d_in[1];
  const float* c1b  = (const float*)d_in[2];
  const float* c2w  = (const float*)d_in[3];
  const float* c2b  = (const float*)d_in[4];
  const float* damp = (const float*)d_in[5];
  const float* sw   = (const float*)d_in[6];
  const float* sb   = (const float*)d_in[7];
  const float* g1w  = (const float*)d_in[8];
  const float* g1b  = (const float*)d_in[9];
  const float* g2w  = (const float*)d_in[10];
  const float* g2b  = (const float*)d_in[11];
  const float* gw   = (const float*)d_in[12];
  const float* gb   = (const float*)d_in[13];
  const float* aw1  = (const float*)d_in[14];
  const float* ab1  = (const float*)d_in[15];
  const float* aw2  = (const float*)d_in[16];
  const float* ab2  = (const float*)d_in[17];

  float* out   = (float*)d_out;
  float* cur   = out;                        // ODE state z, then final 'cur'
  float* enh0  = out + 1*(size_t)NTOT;
  float* enh1  = out + 2*(size_t)NTOT;
  float* enh2  = out + 3*(size_t)NTOT;
  float* ortho = out + 4*(size_t)NTOT;
  float* los   = ortho + 1;                  // (3,32,8)
  float* his   = los + 3*B_*FL_;             // (3,32,8)

  float* ws0   = (float*)d_ws;               // needs >= (3*NTOT + 2048) floats
  float* ws1   = ws0 + NTOT;
  float* ws2   = ws1 + NTOT;
  float* means = ws2 + NTOT;                 // B*C

  const dim3 blk(256);
  const int  EW = NTOT / 256;                // 32768 blocks for elementwise
  const float H = ODE_H;

  // z = x
  hipMemcpyAsync(cur, x, sizeof(float)*NTOT, hipMemcpyDeviceToDevice, stream);

  // ---- RK4 neural ODE: 10 steps, 4 fused stages each ----
  for (int s = 0; s < 10; ++s){
    // k1: acc = z + H/6*k1 ; ws1 = z + H/2*k1
    k_f_eval<<<512, blk, FE_LDS, stream>>>(cur, cur, ws0, ws1,
                                           c1w, c1b, c2w, c2b, damp,
                                           H/6.f, H*0.5f, 0);
    // k2: acc += H/3*k2 ; ws2 = z + H/2*k2
    k_f_eval<<<512, blk, FE_LDS, stream>>>(ws1, cur, ws0, ws2,
                                           c1w, c1b, c2w, c2b, damp,
                                           H/3.f, H*0.5f, 1);
    // k3: acc += H/3*k3 ; ws1 = z + H*k3
    k_f_eval<<<512, blk, FE_LDS, stream>>>(ws2, cur, ws0, ws1,
                                           c1w, c1b, c2w, c2b, damp,
                                           H/3.f, H, 1);
    // k4: z = acc + H/6*k4
    k_f_eval<<<512, blk, FE_LDS, stream>>>(ws1, cur, ws0, cur,
                                           c1w, c1b, c2w, c2b, damp,
                                           H/6.f, 0.f, 2);
  }

  // ---- wavelet decomposition: 3 levels ----
  const float* ap_in[3]  = { cur, ws0, ws1 };
  float*       ap_out[3] = { ws0, ws1, ws2 };
  float*       dets[3]   = { enh0, enh1, enh2 };
  for (int lev = 0; lev < 3; ++lev){
    k_mean <<<B_*C_, blk, 0, stream>>>(ap_in[lev], means);
    k_dywan<<<1,     blk, 0, stream>>>(means, sw, sb, g1w, g1b, g2w, g2b,
                                       los + lev*B_*FL_, his + lev*B_*FL_, ortho, lev);
    k_fir  <<<EW,    blk, 0, stream>>>(ap_in[lev], ap_out[lev], dets[lev],
                                       los + lev*B_*FL_, his + lev*B_*FL_);
  }

  // ---- reconstruction: i = 2, 1, 0 (ping-pong cur to avoid halo races) ----
  k_recon<<<512, blk, RC_LDS, stream>>>(cur, ws0, enh2,
                                        gw + 2*64*64*3, gb + 2*64,
                                        aw1, ab1, aw2, ab2, 0);
  k_recon<<<512, blk, RC_LDS, stream>>>(ws0, ws1, enh1,
                                        gw + 1*64*64*3, gb + 1*64,
                                        aw1 + 1*MID_*C_, ab1 + MID_,
                                        aw2 + 1*C_*MID_, ab2 + C_, 1);
  k_recon<<<512, blk, RC_LDS, stream>>>(ws1, cur, enh0,
                                        gw, gb, aw1, ab1, aw2, ab2, 1);
}
